// TransformerDecoderLayer_70935679861303
// MI455X (gfx1250) — matrix operand order repaired and checked by reading, not compiled, not run on hardware
//
#include <hip/hip_runtime.h>
#include <hip/hip_bf16.h>
#include <math.h>

// ---- problem constants (match reference) ----
#define BB   8
#define SS   1024
#define EE   2048
#define HH   16
#define DD   128
#define FF   8192
#define TOK  (BB*SS)      // 8192 tokens
#define EPSLN 1e-5f

// ---- trivial vector types (safe for unions, unlike HIP's int4) ----
typedef int    i4    __attribute__((ext_vector_type(4)));
typedef __bf16 v16bf __attribute__((ext_vector_type(16)));
typedef float  v8f   __attribute__((ext_vector_type(8)));

union FragU { i4 i[2]; v16bf v; };

__device__ __forceinline__ unsigned short f2bf(float f) {
    __hip_bfloat16 h = __float2bfloat16(f);
    unsigned short u; __builtin_memcpy(&u, &h, 2); return u;
}
__device__ __forceinline__ float bf2f(unsigned short u) {
    unsigned int x = ((unsigned int)u) << 16;
    float f; __builtin_memcpy(&f, &x, 4); return f;
}

// 16-bit A-fragment as loaded here (16x32, MxK): lane l holds row m=l%16,
// k = 16*(l/16)..+15 (contiguous) -> two b128 loads.
__device__ __forceinline__ v16bf frag_a(const unsigned short* base, int ld) {
    int l = threadIdx.x & 31;
    const unsigned short* p = base + (size_t)(l & 15) * ld + ((l >> 4) << 4);
    FragU u; u.i[0] = *(const i4*)p; u.i[1] = *(const i4*)(p + 8);
    return u.v;
}
// 16-bit B-fragment as loaded here (32x16, KxN) from a row-major [N][K] tile:
// lane l holds column n=l%16, k = 16*(l/16)..+15 (contiguous) -> two b128 loads.
__device__ __forceinline__ v16bf frag_b(const unsigned short* base, int ld) {
    int l = threadIdx.x & 31;
    const unsigned short* p = base + (size_t)(l & 15) * ld + ((l >> 4) << 4);
    FragU u; u.i[0] = *(const i4*)p; u.i[1] = *(const i4*)(p + 8);
    return u.v;
}
__device__ __forceinline__ v8f wmma_bf16(v16bf a, v16bf b, v8f c) {
    return __builtin_amdgcn_wmma_f32_16x16x32_bf16(false, a, false, b, (short)0, c, false, false);
}

// ---------------------------------------------------------------------------
// fp32 -> bf16 weight/bias conversion (grid-stride)
// ---------------------------------------------------------------------------
__global__ __launch_bounds__(256)
void k_f32_to_bf16(const float* __restrict__ src, unsigned short* __restrict__ dst, size_t n) {
    for (size_t i = (size_t)blockIdx.x * blockDim.x + threadIdx.x; i < n;
         i += (size_t)gridDim.x * blockDim.x)
        dst[i] = f2bf(src[i]);
}

// ---------------------------------------------------------------------------
// LayerNorm over E=2048, one block per token, bf16 output
// ---------------------------------------------------------------------------
__global__ __launch_bounds__(256)
void k_layernorm_bf16(const float* __restrict__ x, const float* __restrict__ sc,
                      const float* __restrict__ bi, unsigned short* __restrict__ out) {
    __shared__ float r1[256], r2[256];
    const int t = threadIdx.x;
    const size_t base = (size_t)blockIdx.x * EE;
    float v[8], s1 = 0.f, s2 = 0.f;
#pragma unroll
    for (int i = 0; i < 8; i++) {
        v[i] = x[base + t + i * 256];
        s1 += v[i]; s2 += v[i] * v[i];
    }
    r1[t] = s1; r2[t] = s2;
    __syncthreads();
    for (int o = 128; o > 0; o >>= 1) {
        if (t < o) { r1[t] += r1[t + o]; r2[t] += r2[t + o]; }
        __syncthreads();
    }
    const float mu   = r1[0] * (1.f / EE);
    const float var  = r2[0] * (1.f / EE) - mu * mu;
    const float rstd = rsqrtf(var + EPSLN);
#pragma unroll
    for (int i = 0; i < 8; i++) {
        int e = t + i * 256;
        out[base + e] = f2bf((v[i] - mu) * rstd * sc[e] + bi[e]);
    }
}

// ---------------------------------------------------------------------------
// WMMA GEMM: C[M,N] = A[M,K] @ B[N,K]^T (+ epilogue)
// A,B bf16 row-major (K contiguous).  Block tile 128x128, 8 waves, K-step 32.
// MODE 0: +qkv_b, scatter bf16 to [3,B,H,S,D]
// MODE 1: +out_b + resid(x)  -> fp32 [tok,E]
// MODE 2: +gate_b/up_b       -> bf16 [tok,2F]
// MODE 3: +down_b + resid(x1)-> fp32 [tok,E] (d_out)
// ---------------------------------------------------------------------------
template <int MODE>
__global__ __launch_bounds__(256)
void k_gemm_bf16_nt(const unsigned short* __restrict__ A,
                    const unsigned short* __restrict__ Bw,
                    int M, int N, int K,
                    const float* __restrict__ bias0,
                    const float* __restrict__ bias1,
                    const float* __restrict__ resid,
                    unsigned short* __restrict__ out_bf,
                    float* __restrict__ out_f) {
    __shared__ unsigned short As[128 * 32];
    __shared__ unsigned short Bs[128 * 32];
    const int bm = blockIdx.y * 128, bn = blockIdx.x * 128;
    const int t = threadIdx.x, wave = t >> 5, l = t & 31;
    const int wm = (wave >> 2) * 64, wn = (wave & 3) * 32;   // 2x4 wave grid
    v8f acc[4][2] = {};
    const int row = t >> 1, cc = (t & 1) * 16;               // tile staging map
    const unsigned short* ga = A  + (size_t)(bm + row) * K + cc;
    const unsigned short* gb = Bw + (size_t)(bn + row) * K + cc;

    for (int k0 = 0; k0 < K; k0 += 32) {
        i4 a0 = ((const i4*)(ga + k0))[0], a1 = ((const i4*)(ga + k0))[1];
        i4 b0 = ((const i4*)(gb + k0))[0], b1 = ((const i4*)(gb + k0))[1];
        if (k0 + 32 < K) {                       // -> global_prefetch_b8
            __builtin_prefetch(ga + k0 + 32, 0, 0);
            __builtin_prefetch(gb + k0 + 32, 0, 0);
        }
        *(i4*)(As + row * 32 + cc) = a0; *(i4*)(As + row * 32 + cc + 8) = a1;
        *(i4*)(Bs + row * 32 + cc) = b0; *(i4*)(Bs + row * 32 + cc + 8) = b1;
        __syncthreads();
        v16bf bf0 = frag_b(Bs + (wn     ) * 32, 32);
        v16bf bf1 = frag_b(Bs + (wn + 16) * 32, 32);
#pragma unroll
        for (int i = 0; i < 4; i++) {
            v16bf af = frag_a(As + (wm + i * 16) * 32, 32);
            acc[i][0] = wmma_bf16(af, bf0, acc[i][0]);
            acc[i][1] = wmma_bf16(af, bf1, acc[i][1]);
        }
        __syncthreads();
    }

    // C layout: VGPR r holds (m = r + 8*(l/16), n = l%16)
    const int hi = (l >> 4) * 8, cn = l & 15;
#pragma unroll
    for (int i = 0; i < 4; i++)
#pragma unroll
        for (int j = 0; j < 2; j++)
#pragma unroll
            for (int r = 0; r < 8; r++) {
                int grow = bm + wm + i * 16 + hi + r;
                int gcol = bn + wn + j * 16 + cn;
                float v = acc[i][j][r];
                if (MODE == 0) {
                    v += bias0[gcol];
                    int which = gcol >> 11, rem = gcol & 2047;
                    int h = rem >> 7, d = rem & 127;
                    int b = grow >> 10, s = grow & 1023;
                    size_t idx = ((((size_t)which * BB + b) * HH + h) * SS + s) * DD + d;
                    out_bf[idx] = f2bf(v);
                } else if (MODE == 1) {
                    v += bias0[gcol] + resid[(size_t)grow * EE + gcol];
                    out_f[(size_t)grow * EE + gcol] = v;
                } else if (MODE == 2) {
                    v += (gcol < FF) ? bias0[gcol] : bias1[gcol - FF];
                    out_bf[(size_t)grow * (2 * FF) + gcol] = f2bf(v);
                } else {
                    v += bias0[gcol] + resid[(size_t)grow * EE + gcol];
                    out_f[(size_t)grow * EE + gcol] = v;
                }
            }
}

// ---------------------------------------------------------------------------
// Flash attention: block = 4 waves = 64 q-rows for one (b,h); key blocks of 32.
// QK^T and P@V both on v_wmma_f32_16x16x32_bf16; online softmax in fp32.
// qkv layout: [3][B][H][S][D] bf16.  Output: [tok][E] bf16 (head-concat).
// ---------------------------------------------------------------------------
__global__ __launch_bounds__(128)
void k_attn_flash(const unsigned short* __restrict__ qkv,
                  unsigned short* __restrict__ outp) {
    __shared__ unsigned short VT[128][32];        // V^T tile: [d][key]
    __shared__ float          scb[4][16][32];     // raw scores per wave
    __shared__ unsigned short pbuf[4][16][32];    // exp(S-m) bf16 per wave
    __shared__ float          alpha_s[4][16], linv_s[4][16];

    const int qt = blockIdx.x, h = blockIdx.y, b = blockIdx.z;
    const int t = threadIdx.x, w = t >> 5, l = t & 31;
    const size_t plane = (size_t)BB * HH * SS * DD;
    const unsigned short* qp = qkv + ((size_t)b * HH + h) * SS * DD;
    const unsigned short* kp = qp + plane;
    const unsigned short* vp = qp + 2 * plane;
    const int q0 = qt * 64 + w * 16;
    const int hi = (l >> 4) * 8, cn = l & 15;

    // Q fragments for 4 K-chunks of 32 stay resident in VGPRs
    v16bf qf[4];
#pragma unroll
    for (int kk = 0; kk < 4; kk++)
        qf[kk] = frag_a(qp + (size_t)q0 * DD + kk * 32, DD);

    v8f oacc[8] = {};
    float mrow = -3.0e38f, lrow = 0.f;            // valid in lanes 0..15
    const float scale = 0.0883883476483184f;      // 1/sqrt(128)
    const int nkb = qt * 2 + 2;                   // causal key-block count

    for (int kb = 0; kb < nkb; kb++) {
        const int keys = kb * 32;
        // cooperatively stage V^T (32 keys x 128 d) into LDS
        {
            int key = t >> 2, d0 = (t & 3) * 32;
            const unsigned short* vr = vp + (size_t)(keys + key) * DD + d0;
            i4 vv[4];
            vv[0] = ((const i4*)vr)[0]; vv[1] = ((const i4*)vr)[1];
            vv[2] = ((const i4*)vr)[2]; vv[3] = ((const i4*)vr)[3];
            unsigned short tmp[32];
            __builtin_memcpy(tmp, vv, 64);
#pragma unroll
            for (int i = 0; i < 32; i++) VT[d0 + i][key] = tmp[i];
        }
        __syncthreads();

        // scores S = Q K^T for two 16-key subtiles (K frags direct from global)
#pragma unroll
        for (int j = 0; j < 2; j++) {
            v8f sa = {};
#pragma unroll
            for (int kk = 0; kk < 4; kk++) {
                const unsigned short* kbase =
                    kp + (size_t)(keys + j * 16 + cn) * DD + kk * 32 + ((l >> 4) << 4);
                FragU u; u.i[0] = ((const i4*)kbase)[0]; u.i[1] = ((const i4*)kbase)[1];
                sa = wmma_bf16(qf[kk], u.v, sa);
            }
#pragma unroll
            for (int r = 0; r < 8; r++) {
                int qg = q0 + hi + r, kg = keys + j * 16 + cn;
                float sv = sa[r] * scale;
                scb[w][hi + r][j * 16 + cn] = (kg > qg) ? -1.0e30f : sv;
            }
        }
        __builtin_amdgcn_wave_barrier();

        // online-softmax row stats (lanes 0..15 own one row each)
        if (l < 16) {
            float rmax = -3.0e38f;
#pragma unroll
            for (int c = 0; c < 32; c++) rmax = fmaxf(rmax, scb[w][l][c]);
            float mnew = fmaxf(mrow, rmax);
            float al   = __expf(mrow - mnew);
            float ps   = 0.f;
#pragma unroll
            for (int c = 0; c < 32; c++) {
                float p = __expf(scb[w][l][c] - mnew);
                pbuf[w][l][c] = f2bf(p);
                ps += p;
            }
            lrow = al * lrow + ps;
            mrow = mnew;
            alpha_s[w][l] = al;
        }
        __builtin_amdgcn_wave_barrier();

        // rescale O, then O += P @ V
        float a8[8];
#pragma unroll
        for (int r = 0; r < 8; r++) a8[r] = alpha_s[w][hi + r];
#pragma unroll
        for (int dt = 0; dt < 8; dt++)
#pragma unroll
            for (int r = 0; r < 8; r++) oacc[dt][r] *= a8[r];

        v16bf pf = frag_a(&pbuf[w][0][0], 32);
#pragma unroll
        for (int dt = 0; dt < 8; dt++) {
            v16bf vf = frag_b(&VT[dt * 16][0], 32);  // contiguous keys per d-col
            oacc[dt] = wmma_bf16(pf, vf, oacc[dt]);
        }
        __syncthreads();                             // before VT overwrite
    }

    if (l < 16) linv_s[w][l] = 1.f / lrow;
    __builtin_amdgcn_wave_barrier();
    float li8[8];
#pragma unroll
    for (int r = 0; r < 8; r++) li8[r] = linv_s[w][hi + r];
    const size_t trow0 = (size_t)b * SS + q0;
#pragma unroll
    for (int dt = 0; dt < 8; dt++)
#pragma unroll
        for (int r = 0; r < 8; r++) {
            size_t tk = trow0 + hi + r;
            int d = dt * 16 + cn;
            outp[tk * EE + h * DD + d] = f2bf(oacc[dt][r] * li8[r]);
        }
}

// ---------------------------------------------------------------------------
// SwiGLU: act = gelu_exact(gate) * up   (bf16 in/out)
// ---------------------------------------------------------------------------
__global__ __launch_bounds__(256)
void k_swiglu_bf16(const unsigned short* __restrict__ gu,
                   unsigned short* __restrict__ act) {
    const size_t n = (size_t)TOK * FF;
    for (size_t i = (size_t)blockIdx.x * blockDim.x + threadIdx.x; i < n;
         i += (size_t)gridDim.x * blockDim.x) {
        size_t tk = i / FF; int f = (int)(i % FF);
        float g = bf2f(gu[tk * (2 * FF) + f]);
        float u = bf2f(gu[tk * (2 * FF) + FF + f]);
        float ge = 0.5f * g * (1.f + erff(g * 0.70710678118f));
        act[i] = f2bf(ge * u);
    }
}

// ---------------------------------------------------------------------------
extern "C" void kernel_launch(void* const* d_in, const int* in_sizes, int n_in,
                              void* d_out, int out_size, void* d_ws, size_t ws_size,
                              hipStream_t stream) {
    (void)in_sizes; (void)n_in; (void)out_size; (void)ws_size;
    const float* x      = (const float*)d_in[0];
    const float* qkv_w  = (const float*)d_in[1];
    const float* qkv_b  = (const float*)d_in[2];
    const float* out_w  = (const float*)d_in[3];
    const float* out_b  = (const float*)d_in[4];
    const float* ln1_s  = (const float*)d_in[5];
    const float* ln1_b  = (const float*)d_in[6];
    const float* ln2_s  = (const float*)d_in[7];
    const float* ln2_b  = (const float*)d_in[8];
    const float* gate_w = (const float*)d_in[9];
    const float* gate_b = (const float*)d_in[10];
    const float* up_w   = (const float*)d_in[11];
    const float* up_b   = (const float*)d_in[12];
    const float* down_w = (const float*)d_in[13];
    const float* down_b = (const float*)d_in[14];

    // ---- workspace layout (bf16 = 2B elems, fp32 = 4B) ----
    const size_t N_WQKV = (size_t)3 * EE * EE;
    const size_t N_WOUT = (size_t)EE * EE;
    const size_t N_WGU  = (size_t)2 * FF * EE;
    const size_t N_WDN  = (size_t)EE * FF;
    const size_t N_H    = (size_t)TOK * EE;
    const size_t N_QKV  = (size_t)TOK * 3 * EE;
    const size_t N_ATT  = (size_t)TOK * EE;
    const size_t N_X1   = (size_t)TOK * EE;
    const size_t N_GU   = (size_t)TOK * 2 * FF;

    char* p = (char*)d_ws;
    unsigned short* wqkv  = (unsigned short*)p;  p += N_WQKV * 2;
    unsigned short* wout  = (unsigned short*)p;  p += N_WOUT * 2;
    unsigned short* wgu   = (unsigned short*)p;  p += N_WGU  * 2;
    unsigned short* wdown = (unsigned short*)p;  p += N_WDN  * 2;
    unsigned short* hbuf  = (unsigned short*)p;  p += N_H    * 2;
    unsigned short* qkvb  = (unsigned short*)p;  p += N_QKV  * 2;
    unsigned short* attn  = (unsigned short*)p;  p += N_ATT  * 2;
    float*          x1    = (float*)p;           p += N_X1   * 4;
    unsigned short* gub   = (unsigned short*)p;  p += N_GU   * 2;
    unsigned short* actb  = (unsigned short*)p;

    // 1) weights -> bf16 (gate||up packed contiguous as [2F,E])
    k_f32_to_bf16<<<2048, 256, 0, stream>>>(qkv_w,  wqkv,  N_WQKV);
    k_f32_to_bf16<<<2048, 256, 0, stream>>>(out_w,  wout,  N_WOUT);
    k_f32_to_bf16<<<2048, 256, 0, stream>>>(gate_w, wgu,                N_WDN);
    k_f32_to_bf16<<<2048, 256, 0, stream>>>(up_w,   wgu + (size_t)FF*EE, N_WDN);
    k_f32_to_bf16<<<2048, 256, 0, stream>>>(down_w, wdown, N_WDN);

    // 2) h = LN1(x)   (bf16)
    k_layernorm_bf16<<<TOK, 256, 0, stream>>>(x, ln1_s, ln1_b, hbuf);

    // 3) qkv = h @ qkv_w^T + qkv_b   -> [3,B,H,S,D] bf16
    k_gemm_bf16_nt<0><<<dim3(3 * EE / 128, TOK / 128), 256, 0, stream>>>(
        hbuf, wqkv, TOK, 3 * EE, EE, qkv_b, nullptr, nullptr, qkvb, nullptr);

    // 4) causal flash attention -> [tok,E] bf16
    k_attn_flash<<<dim3(SS / 64, HH, BB), 128, 0, stream>>>(qkvb, attn);

    // 5) x1 = attn @ out_w^T + out_b + x   (fp32)
    k_gemm_bf16_nt<1><<<dim3(EE / 128, TOK / 128), 256, 0, stream>>>(
        attn, wout, TOK, EE, EE, out_b, nullptr, x, nullptr, x1);

    // 6) h = LN2(x1)
    k_layernorm_bf16<<<TOK, 256, 0, stream>>>(x1, ln2_s, ln2_b, hbuf);

    // 7) gu = h @ [gate;up]^T + bias    -> [tok,2F] bf16
    k_gemm_bf16_nt<2><<<dim3(2 * FF / 128, TOK / 128), 256, 0, stream>>>(
        hbuf, wgu, TOK, 2 * FF, EE, gate_b, up_b, nullptr, gub, nullptr);

    // 8) act = gelu(gate)*up
    k_swiglu_bf16<<<8192, 256, 0, stream>>>(gub, actb);

    // 9) out = act @ down_w^T + down_b + x1   (fp32 -> d_out)
    k_gemm_bf16_nt<3><<<dim3(EE / 128, TOK / 128), 256, 0, stream>>>(
        actb, wdown, TOK, EE, FF, down_b, nullptr, x1, nullptr, (float*)d_out);
}
